// ContrastiveWSD_25039659335889
// MI455X (gfx1250) — matrix operand order repaired and checked by reading, not compiled, not run on hardware
//
#include <hip/hip_runtime.h>
#include <hip/hip_bf16.h>
#include <stdint.h>

// ---------------- CDNA5 WMMA / TDM types ----------------
typedef __bf16 bf16;
typedef __attribute__((ext_vector_type(16))) __bf16 v16bf;
typedef __attribute__((ext_vector_type(8)))  float  v8f;
typedef __attribute__((ext_vector_type(4)))  unsigned int u32x4;
typedef __attribute__((ext_vector_type(4)))  int i32x4;
typedef __attribute__((ext_vector_type(8)))  int i32x8;

#if __has_builtin(__builtin_amdgcn_tensor_load_to_lds)
#define HAVE_TDM 1
#else
#define HAVE_TDM 0
#endif

// Problem constants (fixed by the reference)
#define BB   64
#define SS   128
#define DD   768
#define HH   256
#define NN   50000
#define EE   500000
#define LL   2048
#define ETOT (EE + NN)

// ---------------- helpers ----------------
__device__ __forceinline__ unsigned fenc(float f) {
    unsigned u = __float_as_uint(f);
    return (u & 0x80000000u) ? ~u : (u | 0x80000000u);
}
__device__ __forceinline__ float fdec(unsigned u) {
    return (u & 0x80000000u) ? __uint_as_float(u & 0x7FFFFFFFu) : __uint_as_float(~u);
}
__device__ __forceinline__ void edge_sd(const int* sp, const int* dp, int e, int& s, int& d) {
    if (e < EE) { s = sp[e]; d = dp[e]; } else { s = d = e - EE; }
}

#if HAVE_TDM
// Issue a 1-D TDM copy of one 16KB B panel (4096 dwords, contiguous) into LDS.
// D# group0: count=1 (user), lds_addr, global_addr[56:0], type=2.
// D# group1: data_size=4B; tensor_dim0 = tile_dim0 = 4096 dwords; stride = 4096.
__device__ __forceinline__ void tdm_load_panel(const bf16* gsrc, unsigned lds_off) {
    unsigned long long ga = (unsigned long long)(uintptr_t)gsrc;
    u32x4 g0;
    g0[0] = 1u;                                                  // count=1, user mode
    g0[1] = lds_off;                                             // lds_addr (bytes)
    g0[2] = (unsigned)(ga & 0xFFFFFFFFu);                        // global_addr[31:0]
    g0[3] = (unsigned)((ga >> 32) & 0x01FFFFFFu) | 0x80000000u;  // addr[56:32] | type=2
    i32x8 g1;
    g1[0] = 0x00020000;              // workgroup_mask=0, data_size=2 (4B)
    g1[1] = (int)(4096u << 16);      // tensor_dim0[15:0]=4096 in [31:16]
    g1[2] = 0;                       // tensor_dim0[31:16]=0, tensor_dim1[15:0]=0
    g1[3] = (int)(4096u << 16);      // tile_dim0 = 4096 dwords
    g1[4] = 0;                       // tile_dim1 = 0 (unused), tile_dim2 = 0
    g1[5] = 4096;                    // tensor_dim0_stride[31:0]
    g1[6] = 0; g1[7] = 0;
    i32x4 z4; z4[0] = 0; z4[1] = 0; z4[2] = 0; z4[3] = 0;
#if __clang_major__ >= 23
    i32x8 z8; z8[0]=0; z8[1]=0; z8[2]=0; z8[3]=0; z8[4]=0; z8[5]=0; z8[6]=0; z8[7]=0;
    __builtin_amdgcn_tensor_load_to_lds(g0, g1, z4, z4, z8, 0);
#else
    __builtin_amdgcn_tensor_load_to_lds(g0, g1, z4, z4, 0);
#endif
}
#endif

// ---------------- weight pack: W[K,256] f32 -> Wp[kb][n][32] bf16 ----------------
// Per 32-K block, all 256 columns' 32 K-values are contiguous: one K-step B panel
// is a single contiguous 16KB chunk (TDM-friendly), and each lane's B fragment
// (column n, K=klo..klo+15) is one contiguous 32B read.
__global__ void pack_w_kernel(const float* __restrict__ W, bf16* __restrict__ Wp, int K) {
    int idx = blockIdx.x * blockDim.x + threadIdx.x;
    int total = K * HH;
    if (idx >= total) return;
    int k = idx / HH, n = idx % HH;
    int kb = k >> 5, kk = k & 31;
    Wp[(((size_t)kb * HH + n) << 5) + kk] = (bf16)W[idx];
}

// ---------------- WMMA GEMM: out[M,256] = X[M,K] @ W[K,256] + bias ----------------
// 256 threads = 8 waves: 4 row strips (16 rows) x 2 col groups (128 cols).
// B panels double-buffered in LDS via TDM (wave 0 issues; s_wait_tensorcnt + barrier).
// A raw floats double-buffered in registers; 8 back-to-back WMMAs per K-step.
__global__ __launch_bounds__(256) void wmma_gemm_kernel(
    const float* __restrict__ X, const bf16* __restrict__ Wp,
    const float* __restrict__ bias, float* __restrict__ out, int M, int K)
{
#if HAVE_TDM
    __shared__ __align__(128) bf16 panel[2][32 * HH];   // 2 x 16KB
#endif
    const int lane = threadIdx.x & 31;
    const int w    = threadIdx.x >> 5;
    const int wr   = w & 3;               // row strip
    const int wc   = w >> 2;              // col group (0/1)
    const int m    = lane & 15;
    const int hi   = lane >> 4;           // 0/1
    const int koff = hi * 16;             // A fragment K sub-offset
    const int klo  = hi * 16;             // B fragment K offset
    const int rowBase = blockIdx.x * 64 + wr * 16;
    int row  = rowBase + m;
    int rowC = row < M ? row : (M - 1);   // clamp loads, guard stores

    v8f acc[8];
#pragma unroll
    for (int t = 0; t < 8; ++t)
#pragma unroll
        for (int r = 0; r < 8; ++r) acc[t][r] = 0.0f;

    const float* xrow = X + (size_t)rowC * K;
    const int nkb = K >> 5;

#if HAVE_TDM
    if (w == 0)
        tdm_load_panel(Wp, (unsigned)(uintptr_t)&panel[0][0]);   // preload panel 0
#endif
    // preload A raw floats for kb=0
    float4 f0 = *(const float4*)(xrow + koff);
    float4 f1 = *(const float4*)(xrow + koff + 4);
    float4 f2 = *(const float4*)(xrow + koff + 8);
    float4 f3 = *(const float4*)(xrow + koff + 12);

    for (int kb = 0; kb < nkb; ++kb) {
#if HAVE_TDM
        if (w == 0) __builtin_amdgcn_s_wait_tensorcnt(0);        // current panel landed
        __syncthreads();                                         // visible to all waves
        if (w == 0 && kb + 1 < nkb)                              // stream next panel
            tdm_load_panel(Wp + ((size_t)(kb + 1) << 13),
                           (unsigned)(uintptr_t)&panel[(kb + 1) & 1][0]);
#endif
        // ---- A fragment (16x32 bf16, contiguous k = 16*hi + s) from prefetched floats ----
        v16bf a;
        a[0]=(bf16)f0.x;  a[1]=(bf16)f0.y;  a[2]=(bf16)f0.z;  a[3]=(bf16)f0.w;
        a[4]=(bf16)f1.x;  a[5]=(bf16)f1.y;  a[6]=(bf16)f1.z;  a[7]=(bf16)f1.w;
        a[8]=(bf16)f2.x;  a[9]=(bf16)f2.y;  a[10]=(bf16)f2.z; a[11]=(bf16)f2.w;
        a[12]=(bf16)f3.x; a[13]=(bf16)f3.y; a[14]=(bf16)f3.z; a[15]=(bf16)f3.w;
        if (kb + 1 < nkb) {                                      // double-buffer A
            const float* nx = xrow + ((kb + 1) << 5);
            f0 = *(const float4*)(nx + koff);
            f1 = *(const float4*)(nx + koff + 4);
            f2 = *(const float4*)(nx + koff + 8);
            f3 = *(const float4*)(nx + koff + 12);
            if (kb + 2 < nkb) __builtin_prefetch(xrow + ((kb + 2) << 5), 0, 3);
        }

        // ---- gather all 8 B fragments, then 8 back-to-back WMMAs ----
        v16bf bfr[8];
#if HAVE_TDM
#pragma unroll
        for (int t = 0; t < 8; ++t) {
            const int n = wc * 128 + t * 16 + m;
            bfr[t] = *(const v16bf*)&panel[kb & 1][(n << 5) + klo];   // ds_load 32B
        }
#else
#pragma unroll
        for (int t = 0; t < 8; ++t) {
            const int n = wc * 128 + t * 16 + m;
            bfr[t] = *(const v16bf*)(Wp + (((size_t)kb * HH + n) << 5) + klo);
        }
#endif
#pragma unroll
        for (int t = 0; t < 8; ++t)
            acc[t] = __builtin_amdgcn_wmma_f32_16x16x32_bf16(
                false, a, false, bfr[t], (short)0, acc[t], false, false);
    }

    // ---- epilogue: D layout row = hi*8 + r, col = m ----
#pragma unroll
    for (int t = 0; t < 8; ++t) {
        const int n = wc * 128 + t * 16 + m;
        const float bv = bias[n];
#pragma unroll
        for (int r = 0; r < 8; ++r) {
            const int rr = rowBase + hi * 8 + r;
            if (rr < M) out[(size_t)rr * HH + n] = acc[t][r] + bv;
        }
    }
}

// ---------------- masked subword sum: refined[b,d] = sum_s mask*agg ----------------
__global__ __launch_bounds__(256) void mask_sum_kernel(
    const float* __restrict__ agg, const int* __restrict__ wid,
    const int* __restrict__ tpos, float* __restrict__ refS)
{
    const int b = blockIdx.x, tid = threadIdx.x;
    const int tp = tpos[b];
    float a0 = 0.f, a1 = 0.f, a2 = 0.f;
    const float* base = agg + (size_t)b * SS * DD;
    const int* wb = wid + b * SS;
    for (int s = 0; s < SS; ++s) {
        if (wb[s] == tp) {
            const float* p = base + (size_t)s * DD;
            a0 += p[tid]; a1 += p[tid + 256]; a2 += p[tid + 512];
        }
    }
    refS[b * DD + tid]       = a0;
    refS[b * DD + tid + 256] = a1;
    refS[b * DD + tid + 512] = a2;
}

// ---------------- LayerNorm + exact GELU over rows of 256 (in-place safe) -------
__global__ __launch_bounds__(256) void norm_gelu_kernel(
    const float* __restrict__ in, const float* __restrict__ g,
    const float* __restrict__ bb, float* __restrict__ out)
{
    const int row = blockIdx.x, tid = threadIdx.x;
    __shared__ float red[256];
    float v = in[(size_t)row * 256 + tid];
    red[tid] = v; __syncthreads();
    for (int s = 128; s > 0; s >>= 1) { if (tid < s) red[tid] += red[tid + s]; __syncthreads(); }
    const float mu = red[0] * (1.0f / 256.0f);
    __syncthreads();
    const float c = v - mu;
    red[tid] = c * c; __syncthreads();
    for (int s = 128; s > 0; s >>= 1) { if (tid < s) red[tid] += red[tid + s]; __syncthreads(); }
    const float var = red[0] * (1.0f / 256.0f);
    const float xn = c * rsqrtf(var + 1e-5f);
    const float y = xn * g[tid] + bb[tid];
    out[(size_t)row * 256 + tid] = 0.5f * y * (1.0f + erff(y * 0.70710678118654752f));
}

// ---------------- init kernels ----------------
__global__ void init_u32_kernel(unsigned* p, unsigned v, int n) {
    int i = blockIdx.x * blockDim.x + threadIdx.x;
    if (i < n) p[i] = v;
}
__global__ void init_bias_rows_kernel(float* p, const float* __restrict__ bias, int total) {
    int i = blockIdx.x * blockDim.x + threadIdx.x;
    if (i < total) p[i] = bias[i & 255];
}

// ---------------- GATv2 edge phase ----------------
__global__ __launch_bounds__(256) void edge_logits_kernel(
    const float* __restrict__ xl, const float* __restrict__ xr,
    const int* __restrict__ sp, const int* __restrict__ dp,
    const float* __restrict__ att, float* __restrict__ logits,
    unsigned* __restrict__ mEnc, int dlim)
{
    const int e = (blockIdx.x * 256 + threadIdx.x) >> 5;
    const int lane = threadIdx.x & 31;
    if (e >= ETOT) return;
    int s, d; edge_sd(sp, dp, e, s, d);
    if (d >= dlim) return;
    const float* pl = xl + (size_t)s * HH;
    const float* pr = xr + (size_t)d * HH;
    float acc = 0.f;
#pragma unroll
    for (int j0 = 0; j0 < HH; j0 += 32) {
        const int j = j0 + lane;
        float v = pl[j] + pr[j];
        v = v > 0.f ? v : 0.2f * v;       // leaky_relu(0.2)
        acc += v * att[j];
    }
    for (int off = 16; off > 0; off >>= 1) acc += __shfl_down(acc, off, 32);
    if (lane == 0) {
        logits[e] = acc;
        atomicMax(&mEnc[d], fenc(acc));   // segment max
    }
}

__global__ __launch_bounds__(256) void edge_exp_kernel(
    const float* __restrict__ logits, const unsigned* __restrict__ mEnc,
    float* __restrict__ den, float* __restrict__ ex,
    const int* __restrict__ sp, const int* __restrict__ dp, int dlim)
{
    const int e = blockIdx.x * 256 + threadIdx.x;
    if (e >= ETOT) return;
    int s, d; edge_sd(sp, dp, e, s, d); (void)s;
    if (d >= dlim) return;
    const float v = __expf(logits[e] - fdec(mEnc[d]));
    ex[e] = v;
    unsafeAtomicAdd(&den[d], v);          // segment sum -> global_atomic_add_f32
}

__global__ __launch_bounds__(256) void edge_aggr_kernel(
    const float* __restrict__ ex, const float* __restrict__ den,
    const float* __restrict__ xl, float* __restrict__ outb,
    const int* __restrict__ sp, const int* __restrict__ dp, int dlim)
{
    const int e = (blockIdx.x * 256 + threadIdx.x) >> 5;
    const int lane = threadIdx.x & 31;
    if (e >= ETOT) return;
    int s, d; edge_sd(sp, dp, e, s, d);
    if (d >= dlim) return;
    const float alpha = ex[e] / den[d];
    const float* pl = xl + (size_t)s * HH;
    float* po = outb + (size_t)d * HH;
#pragma unroll
    for (int j0 = 0; j0 < HH; j0 += 32) {
        const int j = j0 + lane;
        unsafeAtomicAdd(&po[j], alpha * pl[j]);
    }
}

// ---------------- launcher ----------------
extern "C" void kernel_launch(void* const* d_in, const int* in_sizes, int n_in,
                              void* d_out, int out_size, void* d_ws, size_t ws_size,
                              hipStream_t stream) {
    (void)in_sizes; (void)n_in; (void)out_size; (void)ws_size;
    const float* agg    = (const float*)d_in[0];
    const int*   wid    = (const int*)d_in[1];
    const int*   tpos   = (const int*)d_in[2];
    const float* ukc    = (const float*)d_in[3];
    const int*   edges  = (const int*)d_in[4];   // [2,E]
    // d_in[5] = labels_size (device scalar; L fixed at 2048 by reference)
    const float* W_word = (const float*)d_in[6];  const float* b_word = (const float*)d_in[7];
    const float* g1     = (const float*)d_in[8];  const float* be1    = (const float*)d_in[9];
    const float* W_con  = (const float*)d_in[10]; const float* b_con  = (const float*)d_in[11];
    const float* cg1    = (const float*)d_in[12]; const float* cb1    = (const float*)d_in[13];
    const float* Wl2    = (const float*)d_in[14]; const float* bl2    = (const float*)d_in[15];
    const float* Wr2    = (const float*)d_in[16]; const float* br2    = (const float*)d_in[17];
    const float* att2   = (const float*)d_in[18]; const float* bias2  = (const float*)d_in[19];
    const float* cg2    = (const float*)d_in[20]; const float* cb2    = (const float*)d_in[21];
    const float* Wl3    = (const float*)d_in[22]; const float* bl3    = (const float*)d_in[23];
    const float* Wr3    = (const float*)d_in[24]; const float* br3    = (const float*)d_in[25];
    const float* att3   = (const float*)d_in[26]; const float* bias3  = (const float*)d_in[27];
    const float* cg3    = (const float*)d_in[28]; const float* cb3    = (const float*)d_in[29];

    const int* src = edges;
    const int* dst = edges + EE;

    // ---- workspace carve (~211 MB) ----
    char* w = (char*)d_ws;
    auto carve = [&](size_t bytes) { void* p = (void*)w; w += (bytes + 255) & ~(size_t)255; return p; };
    float*    xbuf   = (float*)carve((size_t)NN * HH * 4);  // x1, then layer-3 aggregation target
    float*    xlb    = (float*)carve((size_t)NN * HH * 4);
    float*    xrb    = (float*)carve((size_t)NN * HH * 4);
    float*    goutb  = (float*)carve((size_t)NN * HH * 4);  // layer-2 aggregation target / x2
    float*    refS   = (float*)carve((size_t)BB * DD * 4);
    float*    refPre = (float*)carve((size_t)BB * HH * 4);
    float*    logitsb= (float*)carve((size_t)ETOT * 4);
    float*    exb    = (float*)carve((size_t)ETOT * 4);
    unsigned* mEnc   = (unsigned*)carve((size_t)NN * 4);
    float*    den    = (float*)carve((size_t)NN * 4);
    bf16*     pWord  = (bf16*)carve((size_t)DD * HH * 2);
    bf16*     pCon   = (bf16*)carve((size_t)DD * HH * 2);
    bf16*     pL2    = (bf16*)carve((size_t)HH * HH * 2);
    bf16*     pR2    = (bf16*)carve((size_t)HH * HH * 2);
    bf16*     pL3    = (bf16*)carve((size_t)HH * HH * 2);
    bf16*     pR3    = (bf16*)carve((size_t)HH * HH * 2);

    // ---- weight packing (bf16, WMMA/TDM-friendly) ----
    pack_w_kernel<<<(DD * HH + 255) / 256, 256, 0, stream>>>(W_word, pWord, DD);
    pack_w_kernel<<<(DD * HH + 255) / 256, 256, 0, stream>>>(W_con,  pCon,  DD);
    pack_w_kernel<<<(HH * HH + 255) / 256, 256, 0, stream>>>(Wl2, pL2, HH);
    pack_w_kernel<<<(HH * HH + 255) / 256, 256, 0, stream>>>(Wr2, pR2, HH);
    pack_w_kernel<<<(HH * HH + 255) / 256, 256, 0, stream>>>(Wl3, pL3, HH);
    pack_w_kernel<<<(HH * HH + 255) / 256, 256, 0, stream>>>(Wr3, pR3, HH);

    auto gemm = [&](const float* X, const bf16* Wp, const float* bi, float* O, int M, int K) {
        wmma_gemm_kernel<<<(M + 63) / 64, 256, 0, stream>>>(X, Wp, bi, O, M, K);
    };
    auto gat_edges = [&](const float* xl_, const float* xr_, const float* att_, float* out_, int dlim) {
        init_u32_kernel<<<(NN + 255) / 256, 256, 0, stream>>>(mEnc, 0u, NN);          // enc(-inf) sentinel
        init_u32_kernel<<<(NN + 255) / 256, 256, 0, stream>>>((unsigned*)den, 0u, NN);
        edge_logits_kernel<<<(ETOT * 32 + 255) / 256, 256, 0, stream>>>(xl_, xr_, src, dst, att_, logitsb, mEnc, dlim);
        edge_exp_kernel<<<(ETOT + 255) / 256, 256, 0, stream>>>(logitsb, mEnc, den, exb, src, dst, dlim);
        edge_aggr_kernel<<<(ETOT * 32 + 255) / 256, 256, 0, stream>>>(exb, den, xl_, out_, src, dst, dlim);
    };

    // ---- refined path ----
    mask_sum_kernel<<<BB, 256, 0, stream>>>(agg, wid, tpos, refS);
    gemm(refS, pWord, b_word, refPre, BB, DD);
    norm_gelu_kernel<<<BB, 256, 0, stream>>>(refPre, g1, be1, (float*)d_out);

    // ---- GNN: input projection ----
    gemm(ukc, pCon, b_con, xbuf, NN, DD);
    norm_gelu_kernel<<<NN, 256, 0, stream>>>(xbuf, cg1, cb1, xbuf);     // x1 (in place)

    // ---- GATv2 layer 2 (full graph) ----
    gemm(xbuf, pL2, bl2, xlb, NN, HH);
    gemm(xbuf, pR2, br2, xrb, NN, HH);
    init_bias_rows_kernel<<<((size_t)NN * HH + 255) / 256, 256, 0, stream>>>(goutb, bias2, NN * HH);
    gat_edges(xlb, xrb, att2, goutb, NN);
    norm_gelu_kernel<<<NN, 256, 0, stream>>>(goutb, cg2, cb2, goutb);   // x2 (in place)

    // ---- GATv2 layer 3 (only dst < L=2048 contributes to output) ----
    gemm(goutb, pL3, bl3, xlb, NN, HH);       // xl3 for all src nodes
    gemm(goutb, pR3, br3, xrb, LL, HH);       // xr3 only for dst rows we keep
    init_bias_rows_kernel<<<((size_t)LL * HH + 255) / 256, 256, 0, stream>>>(xbuf, bias3, LL * HH);
    gat_edges(xlb, xrb, att3, xbuf, LL);
    norm_gelu_kernel<<<LL, 256, 0, stream>>>(xbuf, cg3, cb3, (float*)d_out + BB * HH);
}
